// _EmformerLayer_42494406427092
// MI455X (gfx1250) — matrix operand order repaired and checked by reading, not compiled, not run on hardware
//
#include <hip/hip_runtime.h>

// ---------------- problem constants ----------------
#define T_    1024
#define B_    16
#define D_    512
#define H_    8
#define R_    32
#define S_    16
#define M_    16
#define TQ_   1072            // R + T + S
#define KL_   1072            // M + R + T
#define DH_   64              // D / H
#define NROWS 17152           // TQ_*B_ == KL_*B_
#define NEG_INF (-1.0e8f)

typedef __attribute__((ext_vector_type(16))) __bf16     v16bf;
typedef __attribute__((ext_vector_type(8)))  float      v8f;
typedef __attribute__((ext_vector_type(4)))  unsigned   u32x4;
typedef __attribute__((ext_vector_type(4)))  float      f32x4;
typedef __attribute__((ext_vector_type(4)))  unsigned   v4u;
typedef __attribute__((ext_vector_type(4)))  int        v4i;
typedef __attribute__((ext_vector_type(8)))  int        v8i;

// TDM availability (device pass only; host pass takes manual path)
#if defined(__HIP_DEVICE_COMPILE__) && defined(__has_builtin)
#  if __has_builtin(__builtin_amdgcn_tensor_load_to_lds) && __has_builtin(__builtin_amdgcn_s_wait_tensorcnt)
#    define CDNA5_HAS_TDM 1
#  endif
#endif
#ifndef CDNA5_HAS_TDM
#  define CDNA5_HAS_TDM 0
#endif

// 16-bf16 WMMA fragment, viewable as 2x 16-byte chunks or 16 ushorts
union Frag {
  v16bf          v;
  u32x4          u[2];
  unsigned short s[16];
};

__device__ __forceinline__ v8f wmma_bf16(const Frag& a, const Frag& b, v8f c) {
  // D = A(16x32 bf16) * B(32x16 bf16) + C(16x16 f32)
  return __builtin_amdgcn_wmma_f32_16x16x32_bf16(
      /*neg_a=*/false, a.v, /*neg_b=*/false, b.v,
      /*c_mod=*/(short)0, c, /*reuse_a=*/false, /*reuse_b=*/false);
}

__device__ __forceinline__ unsigned short f2bf(float f) {
  union { float f; unsigned u; } x; x.f = f;
  unsigned u = x.u;
  unsigned r = (u + 0x7FFFu + ((u >> 16) & 1u)) >> 16;   // RNE
  return (unsigned short)r;
}

// ---------------- fp32 -> bf16 convert (vector x8) ----------------
__global__ void cvt_f32_bf16(const float* __restrict__ src,
                             unsigned short* __restrict__ dst, long long n) {
  long long i = ((long long)blockIdx.x * blockDim.x + threadIdx.x) * 8;
  if (i >= n) return;
  f32x4 a = *(const f32x4*)(src + i);
  f32x4 b = *(const f32x4*)(src + i + 4);
  u32x4 p;
  p[0] = (unsigned)f2bf(a[0]) | ((unsigned)f2bf(a[1]) << 16);
  p[1] = (unsigned)f2bf(a[2]) | ((unsigned)f2bf(a[3]) << 16);
  p[2] = (unsigned)f2bf(b[0]) | ((unsigned)f2bf(b[1]) << 16);
  p[3] = (unsigned)f2bf(b[2]) | ((unsigned)f2bf(b[3]) << 16);
  *(u32x4*)(dst + i) = p;
}

// ---------------- klengths from lengths (int64) ----------------
__global__ void compute_klen(const long long* __restrict__ lengths,
                             int* __restrict__ klen) {
  if (threadIdx.x == 0) {
    long long mx = 0;
    for (int b = 0; b < B_; ++b) mx = lengths[b] > mx ? lengths[b] : mx;
    int rcbl = TQ_ - (int)mx - S_;
    for (int b = 0; b < B_; ++b) klen[b] = (int)lengths[b] + M_ + rcbl;
  }
}

// ---------------- tiled bf16 WMMA GEMM: C = A(MxK) * W(NxK)^T + bias ----------------
// block tile 128(M) x 128(N); 8 waves in a 4(m) x 2(n) grid; each wave 32x64
// (2 m-tiles x 4 n-tiles = 8 WMMA per 32-deep k-step).
// mode 0: query proj  -> qh[(b*H+h)*TQ + t][d]  (bf16, scaled by 1/8)
// mode 1: kv proj     -> kh / vh head-major bf16
// mode 2: out proj    -> d_out fp32, clip rows >= TQ-S, drop row TQ-1
__global__ __launch_bounds__(256)
void gemm_bf16(const unsigned short* __restrict__ A,
               const unsigned short* __restrict__ W,
               int N, int K, int mode,
               const float* __restrict__ bias,
               unsigned short* __restrict__ ob0,
               unsigned short* __restrict__ ob1,
               float* __restrict__ of) {
  __shared__ unsigned short At[128][32];   // 8 KB
  __shared__ unsigned short Bt[128][32];   // 8 KB
  const int m0   = blockIdx.x * 128;
  const int n0   = blockIdx.y * 128;
  const int tid  = threadIdx.x;
  const int wave = tid >> 5, lane = tid & 31;
  const int lh   = lane >> 4, ln = lane & 15;
  const int mw   = wave >> 1;              // 0..3
  const int nw   = wave & 1;               // 0..1

  v8f acc[2][4];
  #pragma unroll
  for (int mi = 0; mi < 2; ++mi)
    #pragma unroll
    for (int nt = 0; nt < 4; ++nt)
      #pragma unroll
      for (int j = 0; j < 8; ++j) acc[mi][nt][j] = 0.0f;

  for (int k0 = 0; k0 < K; k0 += 32) {
    __syncthreads();
    {   // stage A and W tiles, 128x32 each; 2 threads per row, 16 bf16 each
      int r = tid >> 1, hh = (tid & 1) * 16;
      const u32x4* s = (const u32x4*)(A + (size_t)(m0 + r) * K + k0 + hh);
      *(u32x4*)&At[r][hh]     = s[0];
      *(u32x4*)&At[r][hh + 8] = s[1];
      const u32x4* w = (const u32x4*)(W + (size_t)(n0 + r) * K + k0 + hh);
      *(u32x4*)&Bt[r][hh]     = w[0];
      *(u32x4*)&Bt[r][hh + 8] = w[1];
    }
    __syncthreads();

    // A fragments: lane row = ln, K elems {kb..kb+7, 16+kb..16+kb+7}, kb = lh*8
    Frag af[2];
    #pragma unroll
    for (int mi = 0; mi < 2; ++mi) {
      af[mi].u[0] = *(const u32x4*)&At[mw * 32 + mi * 16 + ln][lh * 8];
      af[mi].u[1] = *(const u32x4*)&At[mw * 32 + mi * 16 + ln][16 + lh * 8];
    }
    #pragma unroll
    for (int nt = 0; nt < 4; ++nt) {
      // B fragment: K elems {kb..kb+7, 16+kb..16+kb+7}, kb = lh*8
      Frag bf;
      bf.u[0] = *(const u32x4*)&Bt[nw * 64 + nt * 16 + ln][lh * 8];
      bf.u[1] = *(const u32x4*)&Bt[nw * 64 + nt * 16 + ln][16 + lh * 8];
      #pragma unroll
      for (int mi = 0; mi < 2; ++mi)
        acc[mi][nt] = wmma_bf16(af[mi], bf, acc[mi][nt]);
    }
  }

  // epilogue: elem j of acc[mi][nt] -> row m0+mw*32+mi*16+lh*8+j, col n0+nw*64+nt*16+ln
  #pragma unroll
  for (int mi = 0; mi < 2; ++mi) {
    #pragma unroll
    for (int nt = 0; nt < 4; ++nt) {
      #pragma unroll
      for (int j = 0; j < 8; ++j) {
        int gm = m0 + mw * 32 + mi * 16 + lh * 8 + j;
        int gn = n0 + nw * 64 + nt * 16 + ln;
        float v = acc[mi][nt][j] + bias[gn];
        int t = gm >> 4, b = gm & 15;          // row = t*B + b
        if (mode == 0) {
          int h = gn >> 6, d = gn & 63;
          ob0[((size_t)(b * H_ + h) * TQ_ + t) * DH_ + d] = f2bf(v * 0.125f);
        } else if (mode == 1) {
          int gn2 = gn & 511;
          int h = gn2 >> 6, d = gn2 & 63;
          unsigned short* dst = (gn < 512) ? ob0 : ob1;
          dst[((size_t)(b * H_ + h) * KL_ + t) * DH_ + d] = f2bf(v);
        } else {
          if (t < TQ_ - 1) {                   // drop last summary row
            if (t >= TQ_ - S_) v = fminf(10.0f, fmaxf(-10.0f, v));
            of[(size_t)gm * D_ + gn] = v;
          }
        }
      }
    }
  }
}

// ---------------- flash attention: one head-batch per block ----------------
// 8 waves x 16 q-rows = 128 q rows per block; kv tiles of 32; online softmax.
// V tile staged into LDS by the Tensor Data Mover when available.
__global__ __launch_bounds__(256)
void flash_attn(const unsigned short* __restrict__ qh,
                const unsigned short* __restrict__ kh,
                const unsigned short* __restrict__ vh,
                const int* __restrict__ klen,
                unsigned short* __restrict__ oin) {
  __shared__ unsigned short Vt[32][64];       // 4 KB (shared V tile)
  __shared__ unsigned short Pt[8][16][48];    // per-wave P staging (padded rows)

  const int n   = blockIdx.y;                 // head-batch = b*H + h
  const int b   = n >> 3;
  const int h   = n & 7;
  const int q0  = blockIdx.x * 128;
  const int tid = threadIdx.x;
  const int wave = tid >> 5, lane = tid & 31;
  const int lh = lane >> 4, ln = lane & 15;
  const int qrow0 = q0 + wave * 16;

  // load Q fragments (pre-scaled bf16), clamp out-of-range rows (never stored)
  Frag qa[2];
  {
    int mr = qrow0 + ln; if (mr > TQ_ - 1) mr = TQ_ - 1;
    const unsigned short* qp = qh + ((size_t)n * TQ_ + mr) * DH_;
    #pragma unroll
    for (int c = 0; c < 2; ++c) {
      qa[c].u[0] = *(const u32x4*)(qp + c * 32 + lh * 8);
      qa[c].u[1] = *(const u32x4*)(qp + c * 32 + 16 + lh * 8);
    }
  }

  float mrow[8], lsum[8];
  #pragma unroll
  for (int j = 0; j < 8; ++j) { mrow[j] = -1.0e30f; lsum[j] = 0.0f; }
  v8f o[4];
  #pragma unroll
  for (int t = 0; t < 4; ++t)
    #pragma unroll
    for (int j = 0; j < 8; ++j) o[t][j] = 0.0f;

  const int klb = klen[b];
  int kend = q0 + 127 + M_ + 1;               // causal: ki <= qi + M
  if (kend > KL_) kend = KL_;
  if (kend > klb) kend = klb;                 // pad mask bound (uniform per block)

  for (int kv0 = 0; kv0 < kend; kv0 += 32) {
    __syncthreads();
#if CDNA5_HAS_TDM
    if (wave == 0) {
      // Tensor DMA: 2-D tile, 32 rows x 64 bf16, row stride 64 elements.
      // tensor_dim1 = remaining rows -> HW zero-fills OOB rows.
      unsigned lds_addr = (unsigned)(unsigned long long)(const void*)&Vt[0][0];
      unsigned long long ga =
          (unsigned long long)(const void*)(vh + ((size_t)n * KL_ + kv0) * DH_);
      int rows = KL_ - kv0; if (rows > 65535) rows = 65535;
      v4u g0;
      g0[0] = 1u;                                   // count=1, user descriptor
      g0[1] = lds_addr;                             // lds_addr[31:0]
      g0[2] = (unsigned)ga;                         // global_addr[31:0]
      g0[3] = ((unsigned)(ga >> 32) & 0x01FFFFFFu)  // global_addr[56:32]
              | 0x80000000u;                        // type=2 ("image")
      v8i g1;
      g1[0] = 0x00010000;          // data_size=1 (2 bytes/elem)
      g1[1] = (64 << 16);          // tensor_dim0[15:0] = 64
      g1[2] = (rows << 16);        // tensor_dim1[15:0] = rows
      g1[3] = (64 << 16);          // tile_dim0 = 64
      g1[4] = 32;                  // tile_dim1 = 32, tile_dim2 = 0
      g1[5] = 64;                  // tensor_dim0_stride low32 = 64
      g1[6] = 0;                   // stride hi / dim1_stride
      g1[7] = 0;
      v4i zg = {0, 0, 0, 0};
#if __clang_major__ >= 23
      v8i zg8 = {0, 0, 0, 0, 0, 0, 0, 0};
      __builtin_amdgcn_tensor_load_to_lds(g0, g1, zg, zg, zg8, 0);
#else
      __builtin_amdgcn_tensor_load_to_lds(g0, g1, zg, zg, 0);
#endif
      __builtin_amdgcn_s_wait_tensorcnt(0);
    }
#else
    {   // manual stage of V tile 32x64
      int r = tid >> 3, cq = (tid & 7) * 8;
      int kr = kv0 + r;
      u32x4 val = {0u, 0u, 0u, 0u};
      if (kr < KL_) val = *(const u32x4*)(vh + ((size_t)n * KL_ + kr) * DH_ + cq);
      *(u32x4*)&Vt[r][cq] = val;
      if (kv0 + 32 < kend)
        __builtin_prefetch(vh + ((size_t)n * KL_ + kv0 + 32 + r) * DH_ + cq, 0, 1);
    }
#endif
    __syncthreads();

    // S = q * k^T for 32 kv cols -> two 16x16 tiles
    v8f s0, s1;
    #pragma unroll
    for (int j = 0; j < 8; ++j) { s0[j] = 0.0f; s1[j] = 0.0f; }
    #pragma unroll
    for (int c = 0; c < 2; ++c) {
      int kr0 = kv0 + ln;      if (kr0 >= KL_) kr0 = KL_ - 1;
      int kr1 = kv0 + 16 + ln; if (kr1 >= KL_) kr1 = KL_ - 1;
      const unsigned short* kp0 = kh + ((size_t)n * KL_ + kr0) * DH_ + c * 32 + lh * 8;
      const unsigned short* kp1 = kh + ((size_t)n * KL_ + kr1) * DH_ + c * 32 + lh * 8;
      Frag kb0, kb1;
      kb0.u[0] = *(const u32x4*)(kp0);
      kb0.u[1] = *(const u32x4*)(kp0 + 16);
      kb1.u[0] = *(const u32x4*)(kp1);
      kb1.u[1] = *(const u32x4*)(kp1 + 16);
      s0 = wmma_bf16(qa[c], kb0, s0);
      s1 = wmma_bf16(qa[c], kb1, s1);
    }

    // masks (causal + padding), analytic
    #pragma unroll
    for (int j = 0; j < 8; ++j) {
      int qi  = qrow0 + lh * 8 + j;
      int ki0 = kv0 + ln;
      int ki1 = kv0 + 16 + ln;
      if (ki0 > qi + M_ || ki0 >= klb) s0[j] = NEG_INF;
      if (ki1 > qi + M_ || ki1 >= klb) s1[j] = NEG_INF;
    }

    // online softmax (row stats shared across 16-lane group)
    float tm[8], rs[8], alpha[8];
    #pragma unroll
    for (int j = 0; j < 8; ++j) tm[j] = fmaxf(s0[j], s1[j]);
    #pragma unroll
    for (int off = 8; off >= 1; off >>= 1)
      #pragma unroll
      for (int j = 0; j < 8; ++j) tm[j] = fmaxf(tm[j], __shfl_xor(tm[j], off, 32));
    #pragma unroll
    for (int j = 0; j < 8; ++j) {
      float nm = fmaxf(mrow[j], tm[j]);
      alpha[j] = __expf(mrow[j] - nm);
      mrow[j]  = nm;
      s0[j] = __expf(s0[j] - nm);
      s1[j] = __expf(s1[j] - nm);
      rs[j] = s0[j] + s1[j];
    }
    #pragma unroll
    for (int off = 8; off >= 1; off >>= 1)
      #pragma unroll
      for (int j = 0; j < 8; ++j) rs[j] += __shfl_xor(rs[j], off, 32);
    #pragma unroll
    for (int j = 0; j < 8; ++j) lsum[j] = lsum[j] * alpha[j] + rs[j];
    #pragma unroll
    for (int t = 0; t < 4; ++t)
      #pragma unroll
      for (int j = 0; j < 8; ++j) o[t][j] *= alpha[j];

    // P: C/D layout -> A layout via per-wave LDS staging
    #pragma unroll
    for (int j = 0; j < 8; ++j) {
      Pt[wave][lh * 8 + j][ln]      = f2bf(s0[j]);
      Pt[wave][lh * 8 + j][16 + ln] = f2bf(s1[j]);
    }
    Frag pa;   // lane row = ln, K elems {kb..kb+7, 16+kb..}, kb = lh*8
    pa.u[0] = *(const u32x4*)&Pt[wave][ln][lh * 8];
    pa.u[1] = *(const u32x4*)&Pt[wave][ln][16 + lh * 8];

    // O += P * V
    #pragma unroll
    for (int dt = 0; dt < 4; ++dt) {
      Frag vb;   // K elems {kb..kb+7, 16+kb..}, kb = lh*8
      #pragma unroll
      for (int i = 0; i < 16; ++i) vb.s[i] = Vt[(i < 8 ? lh * 8 + i : 16 + lh * 8 + (i - 8))][dt * 16 + ln];
      o[dt] = wmma_bf16(pa, vb, o[dt]);
    }
  }

  // epilogue: attn[(b*H+h)][qi][d] -> oin[qi*B + b][h*64 + d] (bf16)
  #pragma unroll
  for (int dt = 0; dt < 4; ++dt) {
    #pragma unroll
    for (int j = 0; j < 8; ++j) {
      int qi = qrow0 + lh * 8 + j;
      if (qi < TQ_) {
        float v = o[dt][j] / lsum[j];
        oin[((size_t)qi * B_ + b) * D_ + h * DH_ + dt * 16 + ln] = f2bf(v);
      }
    }
  }
}

// ---------------- host-side orchestration ----------------
extern "C" void kernel_launch(void* const* d_in, const int* in_sizes, int n_in,
                              void* d_out, int out_size, void* d_ws, size_t ws_size,
                              hipStream_t stream) {
  (void)in_sizes; (void)n_in; (void)out_size; (void)ws_size;
  const float*     utt     = (const float*)d_in[0];
  const long long* lengths = (const long long*)d_in[1];   // int64 per reference
  const float*     rc      = (const float*)d_in[2];
  const float*     summ    = (const float*)d_in[3];
  const float*     mems    = (const float*)d_in[4];
  // d_in[5] attention_mask is recomputed analytically on device
  const float*     Wq      = (const float*)d_in[6];
  const float*     bq      = (const float*)d_in[7];
  const float*     Wkv     = (const float*)d_in[8];
  const float*     bkv     = (const float*)d_in[9];
  const float*     Wo      = (const float*)d_in[10];
  const float*     bo      = (const float*)d_in[11];
  float* out = (float*)d_out;

  char* ws = (char*)d_ws;
  size_t off = 0;
  auto alloc = [&](size_t bytes) -> char* {
    char* p = ws + off;
    off += (bytes + 255) & ~(size_t)255;
    return p;
  };
  const size_t rowsD = (size_t)NROWS * D_;
  unsigned short* Xq   = (unsigned short*)alloc(rowsD * 2);
  unsigned short* Xkv  = (unsigned short*)alloc(rowsD * 2);
  unsigned short* Wqb  = (unsigned short*)alloc((size_t)D_ * D_ * 2);
  unsigned short* Wkvb = (unsigned short*)alloc((size_t)2 * D_ * D_ * 2);
  unsigned short* Wob  = (unsigned short*)alloc((size_t)D_ * D_ * 2);
  unsigned short* qhb  = (unsigned short*)alloc(rowsD * 2);
  unsigned short* khb  = (unsigned short*)alloc(rowsD * 2);
  unsigned short* vhb  = (unsigned short*)alloc(rowsD * 2);
  unsigned short* oin  = (unsigned short*)alloc(rowsD * 2);
  int*            klen = (int*)alloc(256);

  auto cvt = [&](const float* s, unsigned short* dpt, long long nels) {
    int blocks = (int)((nels / 8 + 255) / 256);
    hipLaunchKernelGGL(cvt_f32_bf16, dim3(blocks), dim3(256), 0, stream, s, dpt, nels);
  };

  const long long RB  = (long long)R_ * B_ * D_;   // 262144
  const long long TB  = (long long)T_ * B_ * D_;   // 8388608
  const long long SB  = (long long)S_ * B_ * D_;   // 131072
  const long long MBx = (long long)M_ * B_ * D_;   // 131072

  // concatenations are contiguous along axis 0 -> flat converts
  cvt(rc,   Xq,             RB);
  cvt(utt,  Xq + RB,        TB);
  cvt(summ, Xq + RB + TB,   SB);
  cvt(mems, Xkv,            MBx);
  cvt(rc,   Xkv + MBx,      RB);
  cvt(utt,  Xkv + MBx + RB, TB);
  cvt(Wq,  Wqb,  (long long)D_ * D_);
  cvt(Wkv, Wkvb, (long long)2 * D_ * D_);
  cvt(Wo,  Wob,  (long long)D_ * D_);

  hipLaunchKernelGGL(compute_klen, dim3(1), dim3(32), 0, stream, lengths, klen);

  // query projection (scaled into head-major bf16): N=512 -> 4 n-blocks of 128
  hipLaunchKernelGGL(gemm_bf16, dim3(NROWS / 128, D_ / 128), dim3(256), 0, stream,
                     Xq, Wqb, D_, D_, 0, bq, qhb, (unsigned short*)nullptr, (float*)nullptr);
  // kv projection (split into k / v head-major bf16): N=1024 -> 8 n-blocks
  hipLaunchKernelGGL(gemm_bf16, dim3(NROWS / 128, (2 * D_) / 128), dim3(256), 0, stream,
                     Xkv, Wkvb, 2 * D_, D_, 1, bkv, khb, vhb, (float*)nullptr);
  // attention
  hipLaunchKernelGGL(flash_attn, dim3((TQ_ + 127) / 128, B_ * H_), dim3(256), 0, stream,
                     qhb, khb, vhb, klen, oin);
  // output projection + bias + clip + drop last row
  hipLaunchKernelGGL(gemm_bf16, dim3(NROWS / 128, D_ / 128), dim3(256), 0, stream,
                     oin, Wob, D_, D_, 2, bo, (unsigned short*)nullptr, (unsigned short*)nullptr, out);
}
